// NoisyDense_32744830664989
// MI455X (gfx1250) — matrix operand order repaired and checked by reading, not compiled, not run on hardware
//
#include <hip/hip_runtime.h>

typedef __attribute__((ext_vector_type(16))) __bf16 v16bf;
typedef __attribute__((ext_vector_type(8)))  float  v8f;

#define BLK_M 128
#define BLK_N 128
#define BLK_K 32
#define LDSTR 40                 // padded LDS row stride (bf16 elems): 80B = 20 banks -> conflict-free b128 frags
#define REGION (128 * LDSTR)     // one matrix-half region
#define STG (4 * REGION)         // per-stage: A_hi, A_lo, B_hi, B_lo
#define AH_OFF 0
#define AL_OFF REGION
#define BH_OFF (2 * REGION)
#define BL_OFF (3 * REGION)

union Frag32 { uint4 q[2]; v16bf v; };
union Pack4  { __bf16 h[4];  uint2 q;    };
union Pack16 { __bf16 h[16]; uint4 q[2]; };

__global__ __launch_bounds__(256)
void noisy_dense_wmma(const float* __restrict__ X,
                      const float* __restrict__ W,
                      const float* __restrict__ NMEAN,
                      const float* __restrict__ NSTD,
                      const float* __restrict__ NEPS,
                      float* __restrict__ OUT,
                      int M, int K, int N)
{
    __shared__ __bf16 sh[2 * STG];   // 80 KB double-buffered stage (CDNA5: up to 320KB/WG)

    const int tid  = threadIdx.x;
    const int lane = tid & 31;
    const int wid  = tid >> 5;        // 8 waves
    const int wm   = wid & 1;         // 2 waves along M (64 rows each)
    const int wn   = wid >> 1;        // 4 waves along N (32 cols each)
    const int l16  = lane & 15;
    const int half = lane >> 4;

    const int m0 = blockIdx.y * BLK_M;
    const int n0 = blockIdx.x * BLK_N;

    // ---- global->register staging assignments ----
    const int arow = tid >> 3;            // A: 128 rows x 8 float4 per row
    const int acol = (tid & 7) << 2;
    const int bn   = tid & 127;           // B: thread owns col n0+bn, 16 consecutive k
    const int bk   = (tid >> 7) << 4;

    float4 ar[4];
    float  br[16];

    auto g_loadA = [&](int kt) {
        const float* p = X + (size_t)(m0 + arow) * K + (size_t)kt * BLK_K + acol;
#pragma unroll
        for (int i = 0; i < 4; ++i)
            ar[i] = *(const float4*)(p + (size_t)i * 32 * K);
    };
    auto g_loadB = [&](int kt) {
        const float* p = W + (size_t)(kt * BLK_K + bk) * N + n0 + bn;
#pragma unroll
        for (int j = 0; j < 16; ++j)
            br[j] = p[(size_t)j * N];
    };
    auto s_store = [&](int st) {
        __bf16* s = sh + st * STG;
#pragma unroll
        for (int i = 0; i < 4; ++i) {
            const int r = arow + 32 * i;
            float v[4] = {ar[i].x, ar[i].y, ar[i].z, ar[i].w};
            Pack4 h, l;
#pragma unroll
            for (int j = 0; j < 4; ++j) {
                h.h[j] = (__bf16)v[j];
                l.h[j] = (__bf16)(v[j] - (float)h.h[j]);
            }
            *(uint2*)&s[AH_OFF + r * LDSTR + acol] = h.q;
            *(uint2*)&s[AL_OFF + r * LDSTR + acol] = l.q;
        }
        Pack16 h, l;
#pragma unroll
        for (int j = 0; j < 16; ++j) {
            h.h[j] = (__bf16)br[j];
            l.h[j] = (__bf16)(br[j] - (float)h.h[j]);
        }
        *(uint4*)&s[BH_OFF + bn * LDSTR + bk]     = h.q[0];
        *(uint4*)&s[BH_OFF + bn * LDSTR + bk + 8] = h.q[1];
        *(uint4*)&s[BL_OFF + bn * LDSTR + bk]     = l.q[0];
        *(uint4*)&s[BL_OFF + bn * LDSTR + bk + 8] = l.q[1];
    };

    // ---- LDS->fragment loads matching ISA 16-bit A / B VGPR layouts ----
    auto ld_a = [&](int st, int mi, int off) -> v16bf {
        const int r = wm * 64 + mi * 16 + l16;       // lane = row M
        const __bf16* base = &sh[st * STG + off + r * LDSTR];
        Frag32 f;
        f.q[0] = *(const uint4*)(base + 8 * half);        // K = 8*half .. +7
        f.q[1] = *(const uint4*)(base + 16 + 8 * half);   // K = 16+8*half .. +7
        return f.v;
    };
    auto ld_b = [&](int st, int ni, int off) -> v16bf {
        const int c = wn * 32 + ni * 16 + l16;       // lane = col N
        const __bf16* base = &sh[st * STG + off + c * LDSTR];
        Frag32 f;
        f.q[0] = *(const uint4*)(base + 8 * half);        // K = 8*half .. +7
        f.q[1] = *(const uint4*)(base + 16 + 8 * half);   // K = 16+8*half .. +7
        return f.v;
    };

    v8f acc[4][2];
#pragma unroll
    for (int mi = 0; mi < 4; ++mi)
#pragma unroll
        for (int ni = 0; ni < 2; ++ni)
            acc[mi][ni] = (v8f){};

    auto compute = [&](int st) {
        v16bf ah[4], al[4], bh[2], bl[2];
#pragma unroll
        for (int mi = 0; mi < 4; ++mi) { ah[mi] = ld_a(st, mi, AH_OFF); al[mi] = ld_a(st, mi, AL_OFF); }
#pragma unroll
        for (int ni = 0; ni < 2; ++ni) { bh[ni] = ld_b(st, ni, BH_OFF); bl[ni] = ld_b(st, ni, BL_OFF); }
#pragma unroll
        for (int mi = 0; mi < 4; ++mi)
#pragma unroll
            for (int ni = 0; ni < 2; ++ni) {
                acc[mi][ni] = __builtin_amdgcn_wmma_f32_16x16x32_bf16(
                    false, ah[mi], false, bh[ni], (short)0, acc[mi][ni], false, false);
                acc[mi][ni] = __builtin_amdgcn_wmma_f32_16x16x32_bf16(
                    false, ah[mi], false, bl[ni], (short)0, acc[mi][ni], false, false);
                acc[mi][ni] = __builtin_amdgcn_wmma_f32_16x16x32_bf16(
                    false, al[mi], false, bh[ni], (short)0, acc[mi][ni], false, false);
            }
    };

    // ---- software-pipelined main loop: ONE barrier per iteration (double buffer) ----
    g_loadA(0); g_loadB(0);
    s_store(0);
    const int KT = K / BLK_K;
    for (int kt = 0; kt < KT; ++kt) {
        __syncthreads();                       // stage (kt&1) visible to all waves
        const int cur = kt & 1;
        if (kt + 1 < KT) { g_loadA(kt + 1); g_loadB(kt + 1); }   // prefetch next tile
        compute(cur);
        if (kt + 1 < KT) s_store(cur ^ 1);     // fill other buffer, overlaps with WMMA
    }

    // ---- fused noise epilogue (C/D layout: VGPR i, lane l -> M = i+8*(l>=16), N = l%16) ----
    const float SC = 1.0f / 4096.0f;
#pragma unroll
    for (int ni = 0; ni < 2; ++ni) {
        const int n = n0 + wn * 32 + ni * 16 + l16;
        const float mean = NMEAN[n];
        const float sd   = NSTD[n];
#pragma unroll
        for (int mi = 0; mi < 4; ++mi) {
            const int mbase = m0 + wm * 64 + mi * 16 + 8 * half;
#pragma unroll
            for (int i = 0; i < 8; ++i) {
                const size_t off = (size_t)(mbase + i) * N + n;
                OUT[off] = acc[mi][ni][i] + (mean + sd * NEPS[off]) * SC;
            }
        }
    }
}

extern "C" void kernel_launch(void* const* d_in, const int* in_sizes, int n_in,
                              void* d_out, int out_size, void* d_ws, size_t ws_size,
                              hipStream_t stream) {
    const float* x  = (const float*)d_in[0];
    const float* w  = (const float*)d_in[1];
    const float* nm = (const float*)d_in[2];
    const float* ns = (const float*)d_in[3];
    const float* ne = (const float*)d_in[4];
    float* out = (float*)d_out;

    const int U = in_sizes[2];              // UNITS
    const int D = in_sizes[1] / U;          // D_IN
    const int B = in_sizes[0] / D;          // batch

    dim3 grid(U / BLK_N, B / BLK_M);
    noisy_dense_wmma<<<grid, dim3(256), 0, stream>>>(x, w, nm, ns, ne, out, B, D, U);
}
